// Head_33535104647190
// MI455X (gfx1250) — compile-verified, matrix operand order checked by reading, not run on hardware
//
#include <hip/hip_runtime.h>

typedef __attribute__((ext_vector_type(16))) _Float16 v16h;
typedef __attribute__((ext_vector_type(8)))  _Float16 v8h;
typedef __attribute__((ext_vector_type(8)))  float    v8f;
typedef __attribute__((ext_vector_type(4)))  float    v4f;

#define WMMA_F16(a, b, c) \
  __builtin_amdgcn_wmma_f32_16x16x32_f16(false, (a), false, (b), (short)0, (c), false, false)

static constexpr int TT  = 2048;   // sequence length
static constexpr int BB  = 32;     // batch
static constexpr int CE  = 256;    // n_embd / head size
static constexpr int MM  = BB * TT;          // 65536 rows
static constexpr int HSC = 192;

// LDS tile geometry (halves), padded to spread banks
static constexpr int KP  = 264;    // K tile row stride   (32 rows x 256 cols)
static constexpr int VP  = 40;     // V^T tile row stride (256 rows x 32 cols)
static constexpr int oK0 = 0;
static constexpr int oK1 = 32 * KP;
static constexpr int oV0 = 2 * 32 * KP;
static constexpr int oV1 = oV0 + 256 * VP;
static constexpr int SMEM_HALVES = oV1 + 256 * VP;   // 37376 halves = 74752 B

// ---------------------------------------------------------------------------
// Operand loaders (CDNA5 16-bit A/B layout: lane%16 = row, lane/16 selects the
// K half-split; per lane two contiguous 8-element chunks at k0 and k0+16).
// ---------------------------------------------------------------------------
__device__ __forceinline__ v16h load16_f16(const _Float16* p, int stride,
                                           int rowbase, int kbase, int lane) {
  int r  = rowbase + (lane & 15);
  int k0 = kbase + ((lane >> 4) << 3);
  const _Float16* q = p + (size_t)r * stride + k0;
  v8h lo = *(const v8h*)(q);
  v8h hi = *(const v8h*)(q + 16);
  v16h o;
#pragma unroll
  for (int i = 0; i < 8; ++i) { o[i] = lo[i]; o[8 + i] = hi[i]; }
  return o;
}

// LDS variant (int addressing so ds_load_b128 is generated)
__device__ __forceinline__ v16h lds16(const _Float16* p, int stride,
                                      int rowbase, int kbase, int lane) {
  int r  = rowbase + (lane & 15);
  int k0 = kbase + ((lane >> 4) << 3);
  const _Float16* q = p + r * stride + k0;
  v8h lo = *(const v8h*)(q);
  v8h hi = *(const v8h*)(q + 16);
  v16h o;
#pragma unroll
  for (int i = 0; i < 8; ++i) { o[i] = lo[i]; o[8 + i] = hi[i]; }
  return o;
}

__device__ __forceinline__ v16h load16_f32(const float* p, int stride,
                                           int rowbase, int kbase, int lane) {
  int r  = rowbase + (lane & 15);
  int k0 = kbase + ((lane >> 4) << 3);
  const float* q = p + (size_t)r * stride + k0;
  v4f a0 = *(const v4f*)(q);
  v4f a1 = *(const v4f*)(q + 4);
  v4f a2 = *(const v4f*)(q + 16);
  v4f a3 = *(const v4f*)(q + 20);
  v16h o;
#pragma unroll
  for (int i = 0; i < 4; ++i) {
    o[i]      = (_Float16)a0[i];
    o[4 + i]  = (_Float16)a1[i];
    o[8 + i]  = (_Float16)a2[i];
    o[12 + i] = (_Float16)a3[i];
  }
  return o;
}

// ---------------------------------------------------------------------------
// CDNA5 async global -> LDS copy (64 B per call), tracked with ASYNCcnt.
// INST_OFFSET is added to BOTH the LDS and the global address (ISA 08, 4.4).
// ---------------------------------------------------------------------------
__device__ __forceinline__ void async_copy64(unsigned lds, const _Float16* g) {
  asm volatile(
      "global_load_async_to_lds_b128 %0, %1, off\n\t"
      "global_load_async_to_lds_b128 %0, %1, off offset:16\n\t"
      "global_load_async_to_lds_b128 %0, %1, off offset:32\n\t"
      "global_load_async_to_lds_b128 %0, %1, off offset:48"
      :: "v"(lds), "v"(g) : "memory");
}

__device__ __forceinline__ void wait_async() {
  asm volatile("s_wait_asynccnt 0" ::: "memory");
}

// ---------------------------------------------------------------------------
// Weight transpose+convert: W[K,N] fp32 -> WT[N,K] f16
// ---------------------------------------------------------------------------
__global__ void wt_kernel(const float* __restrict__ W, _Float16* __restrict__ WT,
                          int K, int N) {
  int idx = blockIdx.x * 256 + threadIdx.x;
  if (idx >= K * N) return;
  int k = idx / N, n = idx % N;
  WT[(size_t)n * K + k] = (_Float16)W[idx];
}

// ---------------------------------------------------------------------------
// K1: c_q = x @ Wq_comp ; c_kv = x @ Wkv_comp   (f16 outputs)
// ---------------------------------------------------------------------------
__global__ void proj_c_kernel(const float* __restrict__ x,
                              const _Float16* __restrict__ WTq,
                              const _Float16* __restrict__ WTkv,
                              _Float16* __restrict__ cq, _Float16* __restrict__ ckv) {
  int wave = blockIdx.x * 8 + (threadIdx.x >> 5);
  int lane = threadIdx.x & 31;
  int rowbase = wave * 16;
  int half = lane >> 4, col16 = lane & 15;

  v16h a[8];
#pragma unroll
  for (int c8 = 0; c8 < 8; ++c8) a[c8] = load16_f32(x, CE, rowbase, c8 * 32, lane);

  for (int w = 0; w < 2; ++w) {
    const _Float16* WT = w ? WTkv : WTq;
    _Float16* dst = w ? ckv : cq;
    for (int nt = 0; nt < 16; ++nt) {
      v8f acc = {};
#pragma unroll
      for (int c8 = 0; c8 < 8; ++c8) {
        v16h b = load16_f16(WT, CE, nt * 16, c8 * 32, lane);
        acc = WMMA_F16(a[c8], b, acc);
      }
      int col = nt * 16 + col16;
#pragma unroll
      for (int r = 0; r < 8; ++r)
        dst[(size_t)(rowbase + r + 8 * half) * CE + col] = (_Float16)acc[r];
    }
  }
}

// RoPE applied in C-layout: columns sit on adjacent lanes -> shfl_xor(1) pairing
__device__ __forceinline__ void rope_tile(v8f& acc, int nt, int rowbase,
                                          int lane, int half) {
  int nloc = nt * 16 + (lane & 15);
  int i    = nloc >> 1;
  int par  = lane & 1;
  float theta = __powf(10000.f, -(float)i * (1.0f / 32.0f));
#pragma unroll
  for (int r = 0; r < 8; ++r) {
    int t = (rowbase + r + 8 * half) & (TT - 1);
    float ang = (float)t * theta;
    float s, c;
    __sincosf(ang, &s, &c);
    float val = acc[r];
    float part = __shfl_xor(val, 1, 32);
    acc[r] = par ? (part * s + val * c) : (val * c - part * s);
  }
}

// ---------------------------------------------------------------------------
// K2: q = [c_q@Wqc, rope(c_q@Wqr)]  -> qf16 [M,256]
// ---------------------------------------------------------------------------
__global__ void proj_q_kernel(const _Float16* __restrict__ cq,
                              const _Float16* __restrict__ WTqc,
                              const _Float16* __restrict__ WTqr,
                              _Float16* __restrict__ qf) {
  int wave = blockIdx.x * 8 + (threadIdx.x >> 5);
  int lane = threadIdx.x & 31;
  int rowbase = wave * 16;
  int half = lane >> 4, col16 = lane & 15;

  v16h a[8];
#pragma unroll
  for (int c8 = 0; c8 < 8; ++c8) a[c8] = load16_f16(cq, CE, rowbase, c8 * 32, lane);

  for (int nt = 0; nt < 12; ++nt) {
    v8f acc = {};
#pragma unroll
    for (int c8 = 0; c8 < 8; ++c8)
      acc = WMMA_F16(a[c8], load16_f16(WTqc, CE, nt * 16, c8 * 32, lane), acc);
    int col = nt * 16 + col16;
#pragma unroll
    for (int r = 0; r < 8; ++r)
      qf[(size_t)(rowbase + r + 8 * half) * CE + col] = (_Float16)acc[r];
  }
  for (int nt = 0; nt < 4; ++nt) {
    v8f acc = {};
#pragma unroll
    for (int c8 = 0; c8 < 8; ++c8)
      acc = WMMA_F16(a[c8], load16_f16(WTqr, CE, nt * 16, c8 * 32, lane), acc);
    rope_tile(acc, nt, rowbase, lane, half);
    int col = HSC + nt * 16 + col16;
#pragma unroll
    for (int r = 0; r < 8; ++r)
      qf[(size_t)(rowbase + r + 8 * half) * CE + col] = (_Float16)acc[r];
  }
}

// ---------------------------------------------------------------------------
// K3: k (fp32 out + f16 copy), v (fp32 out + f16 V^T)
// ---------------------------------------------------------------------------
__global__ void proj_kv_kernel(const _Float16* __restrict__ ckv,
                               const _Float16* __restrict__ WTkc,
                               const _Float16* __restrict__ WTkr,
                               const _Float16* __restrict__ WTv,
                               float* __restrict__ kout, float* __restrict__ vout,
                               _Float16* __restrict__ kf, _Float16* __restrict__ vT) {
  int wave = blockIdx.x * 8 + (threadIdx.x >> 5);
  int lane = threadIdx.x & 31;
  int rowbase = wave * 16;
  int half = lane >> 4, col16 = lane & 15;
  int b = rowbase >> 11;

  v16h a[8];
#pragma unroll
  for (int c8 = 0; c8 < 8; ++c8) a[c8] = load16_f16(ckv, CE, rowbase, c8 * 32, lane);

  for (int nt = 0; nt < 12; ++nt) {            // k content
    v8f acc = {};
#pragma unroll
    for (int c8 = 0; c8 < 8; ++c8)
      acc = WMMA_F16(a[c8], load16_f16(WTkc, CE, nt * 16, c8 * 32, lane), acc);
    int col = nt * 16 + col16;
#pragma unroll
    for (int r = 0; r < 8; ++r) {
      size_t idx = (size_t)(rowbase + r + 8 * half) * CE + col;
      kout[idx] = acc[r];
      kf[idx]   = (_Float16)acc[r];
    }
  }
  for (int nt = 0; nt < 4; ++nt) {             // k rope
    v8f acc = {};
#pragma unroll
    for (int c8 = 0; c8 < 8; ++c8)
      acc = WMMA_F16(a[c8], load16_f16(WTkr, CE, nt * 16, c8 * 32, lane), acc);
    rope_tile(acc, nt, rowbase, lane, half);
    int col = HSC + nt * 16 + col16;
#pragma unroll
    for (int r = 0; r < 8; ++r) {
      size_t idx = (size_t)(rowbase + r + 8 * half) * CE + col;
      kout[idx] = acc[r];
      kf[idx]   = (_Float16)acc[r];
    }
  }
  for (int nt = 0; nt < 16; ++nt) {            // v
    v8f acc = {};
#pragma unroll
    for (int c8 = 0; c8 < 8; ++c8)
      acc = WMMA_F16(a[c8], load16_f16(WTv, CE, nt * 16, c8 * 32, lane), acc);
    int col = nt * 16 + col16;
#pragma unroll
    for (int r = 0; r < 8; ++r) {
      int row = rowbase + r + 8 * half;
      vout[(size_t)row * CE + col] = acc[r];
      int t = row & (TT - 1);
      vT[(size_t)b * CE * TT + (size_t)col * TT + t] = (_Float16)acc[r];
    }
  }
}

// ---------------------------------------------------------------------------
// K4: flash attention, LDS-staged. Block = 8 waves = 128 query rows of one
// batch. 32-key K/V tiles double-buffered in LDS via async-to-LDS copies.
// S^T = K·Q^T so softmaxed P lands directly in the next WMMA's A layout.
// ---------------------------------------------------------------------------
__global__ __launch_bounds__(256)
void attn_kernel(const _Float16* __restrict__ qf,
                 const _Float16* __restrict__ kf,
                 const _Float16* __restrict__ vT,
                 float* __restrict__ out) {
  extern __shared__ _Float16 smem[];
  int b    = blockIdx.x >> 4;          // 16 query blocks per batch
  int qblk = (blockIdx.x & 15) * 128;
  int wv   = threadIdx.x >> 5;
  int lane = threadIdx.x & 31;
  int tid  = threadIdx.x;
  int qbase = qblk + wv * 16;
  int half = lane >> 4, qcol = lane & 15;

  const _Float16* kfb = kf + (size_t)b * TT * CE;
  const _Float16* vTb = vT + (size_t)b * CE * TT;

  v16h qB[8];                          // Q^T B-operands (lane = query)
#pragma unroll
  for (int c8 = 0; c8 < 8; ++c8)
    qB[c8] = load16_f16(qf, CE, b * TT + qbase, c8 * 32, lane);

  v8f acc[16];
  v8f zero = {};
#pragma unroll
  for (int i = 0; i < 16; ++i) acc[i] = zero;
  float mq = -1e30f, lq = 0.f;
  const float scale = 0.0625f;         // 1/sqrt(256)

  // cooperative stage of one 32-key tile: K 32x256, V^T 256x32 (f16)
  auto stage = [&](int kbase, int bufK, int bufV) {
    const _Float16* gK = kfb + (size_t)(kbase + (tid >> 3)) * CE + (tid & 7) * 32;
    unsigned lK = (unsigned)(uintptr_t)(smem + bufK + (tid >> 3) * KP + (tid & 7) * 32);
    async_copy64(lK, gK);
    const _Float16* gV = vTb + (size_t)tid * TT + kbase;
    unsigned lV = (unsigned)(uintptr_t)(smem + bufV + tid * VP);
    async_copy64(lV, gV);
  };

  int nkt = (qblk >> 5) + 4;           // key tiles this block needs
  stage(0, oK0, oV0);
  wait_async();
  __syncthreads();

  for (int kt = 0; kt < nkt; ++kt) {
    int cur = kt & 1;
    if (kt + 1 < nkt)
      stage((kt + 1) * 32, cur ? oK0 : oK1, cur ? oV0 : oV1);   // prefetch

    int kbase = kt * 32;
    if (kbase <= qbase + 15) {         // uniform per wave: skip past diagonal
      const _Float16* sK = smem + (cur ? oK1 : oK0);
      const _Float16* sV = smem + (cur ? oV1 : oV0);

      v8f s2[2];
#pragma unroll
      for (int sub = 0; sub < 2; ++sub) {
        s2[sub] = zero;
#pragma unroll
        for (int c8 = 0; c8 < 8; ++c8) {
          v16h aK = lds16(sK, KP, sub * 16, c8 * 32, lane);
          s2[sub] = WMMA_F16(aK, qB[c8], s2[sub]);
        }
      }
      // scale + causal mask + tile max
      float tmax = -1e30f;
#pragma unroll
      for (int sub = 0; sub < 2; ++sub)
#pragma unroll
        for (int r = 0; r < 8; ++r) {
          int kk = kbase + sub * 16 + r + 8 * half;
          float v = s2[sub][r] * scale;
          v = (kk > qbase + qcol) ? -1e30f : v;
          s2[sub][r] = v;
          tmax = fmaxf(tmax, v);
        }
      tmax = fmaxf(tmax, __shfl_xor(tmax, 16, 32));
      float mnew = fmaxf(mq, tmax);
      float corr = __expf(mq - mnew);
      mq = mnew;

      // exp + pack directly into the P A-operand (C-layout == A-layout here)
      float psum = 0.f;
      v16h aP;
#pragma unroll
      for (int sub = 0; sub < 2; ++sub)
#pragma unroll
        for (int r = 0; r < 8; ++r) {
          float e = __expf(s2[sub][r] - mnew);
          psum += e;
          aP[sub * 8 + r] = (_Float16)e;
        }
      psum += __shfl_xor(psum, 16, 32);
      lq = lq * corr + psum;

      float cr[8];
#pragma unroll
      for (int r = 0; r < 8; ++r) cr[r] = __shfl(corr, r + 8 * half, 32);
#pragma unroll
      for (int dt = 0; dt < 16; ++dt)
#pragma unroll
        for (int r = 0; r < 8; ++r) acc[dt][r] *= cr[r];

      // P·V : B-operand rows are V^T rows (lane = d-column, K = key index)
#pragma unroll
      for (int dt = 0; dt < 16; ++dt) {
        v16h bV = lds16(sV, VP, dt * 16, 0, lane);
        acc[dt] = WMMA_F16(aP, bV, acc[dt]);
      }
    }
    wait_async();
    __syncthreads();
  }

  // epilogue: divide by softmax denominator, store fp32
  float lr[8];
#pragma unroll
  for (int r = 0; r < 8; ++r) lr[r] = 1.0f / __shfl(lq, r + 8 * half, 32);
#pragma unroll
  for (int dt = 0; dt < 16; ++dt) {
    int col = dt * 16 + qcol;
#pragma unroll
    for (int r = 0; r < 8; ++r) {
      int row = b * TT + qbase + r + 8 * half;
      out[(size_t)row * CE + col] = acc[dt][r] * lr[r];
    }
  }
}

// ---------------------------------------------------------------------------
extern "C" void kernel_launch(void* const* d_in, const int* in_sizes, int n_in,
                              void* d_out, int out_size, void* d_ws, size_t ws_size,
                              hipStream_t stream) {
  const float* x       = (const float*)d_in[0];
  const float* Wq_comp = (const float*)d_in[1];
  const float* Wkv_comp= (const float*)d_in[2];
  const float* Wqc     = (const float*)d_in[3];
  const float* Wqr     = (const float*)d_in[4];
  const float* Wkc     = (const float*)d_in[5];
  const float* Wkr     = (const float*)d_in[6];
  const float* Wv      = (const float*)d_in[7];

  float* out  = (float*)d_out;
  float* kout = out  + (size_t)MM * CE;
  float* vout = kout + (size_t)MM * CE;

  _Float16* ws = (_Float16*)d_ws;
  size_t off = 0;
  _Float16* WTq  = ws + off; off += 256 * 256;
  _Float16* WTkv = ws + off; off += 256 * 256;
  _Float16* WTqc = ws + off; off += 192 * 256;
  _Float16* WTqr = ws + off; off += 64 * 256;
  _Float16* WTkc = ws + off; off += 192 * 256;
  _Float16* WTkr = ws + off; off += 64 * 256;
  _Float16* WTv  = ws + off; off += 256 * 256;
  _Float16* cq   = ws + off; off += (size_t)MM * CE;
  _Float16* ckv  = ws + off; off += (size_t)MM * CE;
  _Float16* qf   = ws + off; off += (size_t)MM * CE;
  _Float16* vT   = ws + off; off += (size_t)MM * CE;
  _Float16* kfh  = ws + off; off += (size_t)MM * CE;

  auto wt = [&](const float* W, _Float16* WT, int K, int N) {
    int n = K * N;
    wt_kernel<<<(n + 255) / 256, 256, 0, stream>>>(W, WT, K, N);
  };
  wt(Wq_comp,  WTq,  256, 256);
  wt(Wkv_comp, WTkv, 256, 256);
  wt(Wqc,      WTqc, 256, 192);
  wt(Wqr,      WTqr, 256, 64);
  wt(Wkc,      WTkc, 256, 192);
  wt(Wkr,      WTkr, 256, 64);
  wt(Wv,       WTv,  256, 256);

  proj_c_kernel <<<MM / 16 / 8, 256, 0, stream>>>(x, WTq, WTkv, cq, ckv);
  proj_q_kernel <<<MM / 16 / 8, 256, 0, stream>>>(cq, WTqc, WTqr, qf);
  proj_kv_kernel<<<MM / 16 / 8, 256, 0, stream>>>(ckv, WTkc, WTkr, WTv,
                                                  kout, vout, kfh, vT);
  attn_kernel<<<BB * (TT / 128), 256, SMEM_HALVES * 2, stream>>>(qf, kfh, vT, out);
}
